// Encoder_68736656605637
// MI455X (gfx1250) — matrix operand order repaired and checked by reading, not compiled, not run on hardware
//
#include <hip/hip_runtime.h>
#include <hip/hip_bf16.h>

// ---------------------------------------------------------------------------
// GPT-style encoder forward for MI455X (gfx1250).
// - WMMA f16 GEMM core (v_wmma_f32_16x16x32_f16), 128x128 block tiles
// - Double-buffered LDS staging via GLOBAL_LOAD_ASYNC_TO_LDS_B128 + asynccnt
// - Weights pre-transposed to [N][K] f16 so all weight GEMMs stream b128 rows
// ---------------------------------------------------------------------------

typedef _Float16 half_t;
typedef __attribute__((ext_vector_type(16))) _Float16 v16h;
typedef __attribute__((ext_vector_type(8)))  float    v8f;
typedef __attribute__((ext_vector_type(4)))  float    f4;

#define BSZ 32
#define TT  512
#define CC  768
#define HH  12
#define LL  12
#define HSZ 64
#define FF  3072
#define MTOK (BSZ * TT)   // 16384

// Low 32 bits of a generic pointer to LDS == LDS byte address (aperture spec).
__device__ __forceinline__ unsigned lds_lo(const void* p) {
    return (unsigned)(unsigned long long)p;
}

// Async global -> LDS copy of 16 bytes (tracked by ASYNCcnt).
__device__ __forceinline__ void async_b128(unsigned lds_off, const half_t* gsrc) {
    asm volatile("global_load_async_to_lds_b128 %0, %1, off"
                 :: "v"(lds_off), "v"(gsrc)
                 : "memory");
}

__device__ __forceinline__ void wait_async0() {
    asm volatile("s_wait_asynccnt 0x0" ::: "memory");
}

// ---------------------------------------------------------------------------
// Transpose + convert weights: src f32 [K][N] (per z-slice) -> dst f16 [N][K]
// ---------------------------------------------------------------------------
__global__ __launch_bounds__(256)
void cvt_tr_f16(const float* __restrict__ s, half_t* __restrict__ d,
                int K, int N) {
    __shared__ float tile[32][33];
    long long slab = (long long)blockIdx.z * K * N;
    s += slab;
    d += slab;
    int k0 = blockIdx.y * 32, n0 = blockIdx.x * 32;
    int tx = threadIdx.x & 31, ty = threadIdx.x >> 5;   // 32 x 8
    for (int i = ty; i < 32; i += 8) {
        int k = k0 + i, n = n0 + tx;
        tile[i][tx] = (k < K && n < N) ? s[(long long)k * N + n] : 0.f;
    }
    __syncthreads();
    for (int i = ty; i < 32; i += 8) {
        int n = n0 + i, k = k0 + tx;
        if (n < N && k < K) d[(long long)n * K + k] = (half_t)tile[tx][i];
    }
}

// ---------------------------------------------------------------------------
// Embedding: x = tok_emb[idx] + pos_emb  (writes f32 master + f16 copy)
// ---------------------------------------------------------------------------
__global__ void embed_kernel(const int* __restrict__ idx,
                             const float* __restrict__ tok,
                             const float* __restrict__ pos,
                             float* __restrict__ x, half_t* __restrict__ x16) {
    int row = blockIdx.x;
    int t = row % TT;
    int token = idx[row];
    const float* te = tok + (long long)token * CC;
    const float* pe = pos + (long long)t * CC;
    long long base = (long long)row * CC;
    for (int c = threadIdx.x; c < CC; c += blockDim.x) {
        float v = te[c] + pe[c];
        x[base + c] = v;
        x16[base + c] = (half_t)v;
    }
}

// ---------------------------------------------------------------------------
// LayerNorm: if add: x = x + LN(y); else x = LN(y). Refreshes x16.
// ---------------------------------------------------------------------------
__global__ __launch_bounds__(256)
void ln_kernel(float* __restrict__ x, half_t* __restrict__ x16,
               const float* __restrict__ y, const float* __restrict__ g,
               const float* __restrict__ b, int add) {
    __shared__ float red[256];
    long long row = blockIdx.x;
    const float* yr = y + row * CC;
    float* xr = x + row * CC;
    half_t* hr = x16 + row * CC;
    int tid = threadIdx.x;

    float s = 0.f;
    for (int c = tid; c < CC; c += 256) s += yr[c];
    red[tid] = s; __syncthreads();
    for (int st = 128; st > 0; st >>= 1) {
        if (tid < st) red[tid] += red[tid + st];
        __syncthreads();
    }
    float mean = red[0] * (1.f / CC);
    __syncthreads();

    float vs = 0.f;
    for (int c = tid; c < CC; c += 256) { float dd = yr[c] - mean; vs += dd * dd; }
    red[tid] = vs; __syncthreads();
    for (int st = 128; st > 0; st >>= 1) {
        if (tid < st) red[tid] += red[tid + st];
        __syncthreads();
    }
    float rstd = rsqrtf(red[0] * (1.f / CC) + 1e-5f);

    for (int c = tid; c < CC; c += 256) {
        float ln = (yr[c] - mean) * rstd * g[c] + b[c];
        float nv = add ? (xr[c] + ln) : ln;
        xr[c] = nv;
        hr[c] = (half_t)nv;
    }
}

// ---------------------------------------------------------------------------
// Softmax over rows of 512 (attention scores), f32 in -> f16 probs out.
// ---------------------------------------------------------------------------
__global__ __launch_bounds__(256)
void softmax_kernel(const float* __restrict__ S, half_t* __restrict__ P) {
    __shared__ float red[256];
    long long row = (long long)blockIdx.x;
    const float* s = S + row * TT;
    half_t* p = P + row * TT;
    int tid = threadIdx.x;

    float v0 = s[tid], v1 = s[tid + 256];
    red[tid] = fmaxf(v0, v1); __syncthreads();
    for (int st = 128; st > 0; st >>= 1) {
        if (tid < st) red[tid] = fmaxf(red[tid], red[tid + st]);
        __syncthreads();
    }
    float m = red[0];
    __syncthreads();
    float e0 = __expf(v0 - m), e1 = __expf(v1 - m);
    red[tid] = e0 + e1; __syncthreads();
    for (int st = 128; st > 0; st >>= 1) {
        if (tid < st) red[tid] += red[tid + st];
        __syncthreads();
    }
    float inv = 1.f / red[0];
    p[tid] = (half_t)(e0 * inv);
    p[tid + 256] = (half_t)(e1 * inv);
}

// ---------------------------------------------------------------------------
// WMMA GEMM: C[M,N] = alpha * A[M,K] x B + bias, optional ReLU.
//   A: f16 row-major (lda). M must be a multiple of 128 (true for all calls).
//   B: bTrans==1 -> stored [N][K] (ldb=K row stride); async-staged; N%128==0.
//      bTrans==0 -> stored [K][N] (ldb=N row stride); scalar-staged; N guarded.
//   Batched over blockIdx.z: off = (z/hb)*_B + (z%hb)*_H (element offsets).
// 128x128 block tile, BK=32, 256 threads = 8 waves (each 32x64), double-
// buffered LDS filled by GLOBAL_LOAD_ASYNC_TO_LDS_B128.
// ---------------------------------------------------------------------------
__global__ __launch_bounds__(256)
void gemm_wmma_f16(const half_t* __restrict__ A, const half_t* __restrict__ B,
                   float* __restrict__ Cf, half_t* __restrict__ Ch,
                   const float* __restrict__ bias,
                   int M, int N, int K, int lda, int ldb, int ldc,
                   long long aB, long long aH, long long bB, long long bH,
                   long long cB, long long cH, int hb,
                   int bTrans, int relu, float alpha) {
    __shared__ half_t lA[2][128 * 40];
    __shared__ half_t lB[2][128 * 40];

    const int tid = threadIdx.x;
    const int lane = tid & 31;
    const int wave = tid >> 5;
    const int waveMoff = (wave & 3) * 32;   // 4 waves along M
    const int waveNoff = (wave >> 2) * 64;  // 2 waves along N
    const int rlo = lane & 15;
    const int halfsel = lane >> 4;

    const int z = blockIdx.z;
    const long long offA = (long long)(z / hb) * aB + (long long)(z % hb) * aH;
    const long long offB = (long long)(z / hb) * bB + (long long)(z % hb) * bH;
    const long long offC = (long long)(z / hb) * cB + (long long)(z % hb) * cH;
    A += offA;
    B += offB;

    const int n0 = blockIdx.x * 128;
    const int m0 = blockIdx.y * 128;

    // Per-thread staging geometry (16 f16 per thread per tile).
    const int e = tid * 16;
    const int arow = e >> 5, akk = e & 31;          // A: [row][k], k in {0,16}
    const half_t* aSrc = A + (long long)(m0 + arow) * lda + akk;
    const unsigned aOff0 = lds_lo(&lA[0][arow * 40 + akk]);
    const unsigned aOff1 = lds_lo(&lA[1][arow * 40 + akk]);
    // B (bTrans): [n][k]
    const half_t* bSrc = B + (long long)(n0 + (e >> 5)) * ldb + (e & 31);
    const unsigned bOff0 = lds_lo(&lB[0][(e >> 5) * 40 + (e & 31)]);
    const unsigned bOff1 = lds_lo(&lB[1][(e >> 5) * 40 + (e & 31)]);
    // B scalar path: [k][n..n+15]
    const int skk = e >> 7, snn = e & 127;

    v8f acc[2][4];
#pragma unroll
    for (int i = 0; i < 2; ++i)
#pragma unroll
        for (int j = 0; j < 4; ++j)
#pragma unroll
            for (int r = 0; r < 8; ++r) acc[i][j][r] = 0.0f;

    const int nstages = K >> 5;   // K % 32 == 0 for all calls

    // ---- issue stage 0 ----
    {
        async_b128(aOff0, aSrc);
        async_b128(aOff0 + 16u, aSrc + 8);
        if (bTrans) {
            async_b128(bOff0, bSrc);
            async_b128(bOff0 + 16u, bSrc + 8);
        } else {
            const long long rowb = (long long)skk * ldb;
#pragma unroll
            for (int i = 0; i < 16; ++i) {
                int gn = n0 + snn + i;
                half_t v = (half_t)0.f;
                if (gn < N) v = B[rowb + gn];
                lB[0][(snn + i) * 40 + skk] = v;
            }
        }
    }

    for (int s = 0; s < nstages; ++s) {
        const int buf = s & 1;
        const int nbuf = buf ^ 1;

        wait_async0();          // this wave's loads for `buf` complete
        __syncthreads();        // everyone's staging for `buf` visible

        // ---- issue stage s+1 into nbuf (overlaps with WMMA below) ----
        if (s + 1 < nstages) {
            const int k0 = (s + 1) << 5;
            const unsigned aO = nbuf ? aOff1 : aOff0;
            async_b128(aO, aSrc + k0);
            async_b128(aO + 16u, aSrc + k0 + 8);
            if (bTrans) {
                const unsigned bO = nbuf ? bOff1 : bOff0;
                async_b128(bO, bSrc + k0);
                async_b128(bO + 16u, bSrc + k0 + 8);
            } else {
                const long long rowb = (long long)(k0 + skk) * ldb;
#pragma unroll
                for (int i = 0; i < 16; ++i) {
                    int gn = n0 + snn + i;
                    half_t v = (half_t)0.f;
                    if (gn < N) v = B[rowb + gn];
                    lB[nbuf][(snn + i) * 40 + skk] = v;
                }
            }
        }

        // ---- fragments + 8 WMMAs on `buf` ----
        v16h af[2];
#pragma unroll
        for (int mt = 0; mt < 2; ++mt) {
            int row = waveMoff + mt * 16 + rlo;
            const half_t* ar0 = &lA[buf][row * 40 + (halfsel << 3)];
            f4* ap = (f4*)&af[mt];
            ap[0] = *(const f4*)(ar0);        // K = hs*8 .. +7
            ap[1] = *(const f4*)(ar0 + 16);   // K = 16 + hs*8 .. +7
        }
        v16h bfr[4];
#pragma unroll
        for (int nt = 0; nt < 4; ++nt) {
            int col = waveNoff + nt * 16 + rlo;
            const half_t* br0 = &lB[buf][col * 40 + (halfsel << 3)];
            f4* bp = (f4*)&bfr[nt];
            bp[0] = *(const f4*)(br0);        // K = hs*8 .. +7
            bp[1] = *(const f4*)(br0 + 16);   // K = 16 + hs*8 .. +7
        }
#pragma unroll
        for (int mt = 0; mt < 2; ++mt)
#pragma unroll
            for (int nt = 0; nt < 4; ++nt)
                acc[mt][nt] = __builtin_amdgcn_wmma_f32_16x16x32_f16(
                    false, af[mt], false, bfr[nt], (short)0, acc[mt][nt],
                    false, false);
        __syncthreads();        // done reading `buf` before it is refilled
    }

    // ---- epilogue: C row = vgpr + 8*halfsel, col = lane%16 ----
    float* Cfp = Cf ? Cf + offC : nullptr;
    half_t* Chp = Ch ? Ch + offC : nullptr;
#pragma unroll
    for (int mt = 0; mt < 2; ++mt)
#pragma unroll
        for (int nt = 0; nt < 4; ++nt) {
            int gn = n0 + waveNoff + nt * 16 + rlo;
            if (gn >= N) continue;
            float bia = bias ? bias[gn] : 0.f;
#pragma unroll
            for (int r = 0; r < 8; ++r) {
                int gm = m0 + waveMoff + mt * 16 + r + (halfsel ? 8 : 0);
                float v = acc[mt][nt][r] * alpha + bia;
                if (relu && v < 0.f) v = 0.f;
                long long ci = (long long)gm * ldc + gn;
                if (Cfp) Cfp[ci] = v;
                if (Chp) Chp[ci] = (half_t)v;
            }
        }
}

// ---------------------------------------------------------------------------
// Mean-pool over T + 4-layer head MLP + sigmoid. One block per batch b.
// ---------------------------------------------------------------------------
__global__ __launch_bounds__(256)
void pool_head_kernel(const float* __restrict__ x,
                      const float* __restrict__ hW1, const float* __restrict__ hb1,
                      const float* __restrict__ hW2, const float* __restrict__ hb2,
                      const float* __restrict__ hW3, const float* __restrict__ hb3,
                      const float* __restrict__ hW4, const float* __restrict__ hb4,
                      float* __restrict__ out) {
    __shared__ float pooled[CC];
    __shared__ float h1[128];
    __shared__ float h2[64];
    __shared__ float h3[16];
    int b = blockIdx.x, tid = threadIdx.x;
    const float* xb = x + (long long)b * TT * CC;
    for (int c = tid; c < CC; c += 256) {
        float s = 0.f;
        for (int t = 0; t < TT; ++t) s += xb[(long long)t * CC + c];
        pooled[c] = s * (1.f / TT);
    }
    __syncthreads();
    for (int j = tid; j < 128; j += 256) {
        float s = hb1[j];
        for (int c = 0; c < CC; ++c) s += pooled[c] * hW1[c * 128 + j];
        h1[j] = fmaxf(s, 0.f);
    }
    __syncthreads();
    if (tid < 64) {
        float s = hb2[tid];
        for (int c = 0; c < 128; ++c) s += h1[c] * hW2[c * 64 + tid];
        h2[tid] = fmaxf(s, 0.f);
    }
    __syncthreads();
    if (tid < 16) {
        float s = hb3[tid];
        for (int c = 0; c < 64; ++c) s += h2[c] * hW3[c * 16 + tid];
        h3[tid] = fmaxf(s, 0.f);
    }
    __syncthreads();
    if (tid < 2) {
        float s = hb4[tid];
        for (int c = 0; c < 16; ++c) s += h3[c] * hW4[c * 2 + tid];
        out[b * 2 + tid] = 1.f / (1.f + __expf(-s));
    }
}

// ---------------------------------------------------------------------------
// Host orchestration
// ---------------------------------------------------------------------------
extern "C" void kernel_launch(void* const* d_in, const int* in_sizes, int n_in,
                              void* d_out, int out_size, void* d_ws, size_t ws_size,
                              hipStream_t stream) {
    (void)in_sizes; (void)n_in; (void)out_size; (void)ws_size;

    const int*   idx   = (const int*)d_in[0];
    const float* tok   = (const float*)d_in[1];
    const float* pos   = (const float*)d_in[2];
    const float* Wq    = (const float*)d_in[3];
    const float* Wk    = (const float*)d_in[4];
    const float* Wv    = (const float*)d_in[5];
    const float* Wo    = (const float*)d_in[6];
    const float* bo    = (const float*)d_in[7];
    const float* ln1g  = (const float*)d_in[8];
    const float* ln1b  = (const float*)d_in[9];
    const float* W1    = (const float*)d_in[10];
    const float* b1    = (const float*)d_in[11];
    const float* W2    = (const float*)d_in[12];
    const float* b2    = (const float*)d_in[13];
    const float* ln2g  = (const float*)d_in[14];
    const float* ln2b  = (const float*)d_in[15];
    const float* lnfg  = (const float*)d_in[16];
    const float* lnfb  = (const float*)d_in[17];
    const float* hW1   = (const float*)d_in[18];
    const float* hb1   = (const float*)d_in[19];
    const float* hW2   = (const float*)d_in[20];
    const float* hb2   = (const float*)d_in[21];
    const float* hW3   = (const float*)d_in[22];
    const float* hb3   = (const float*)d_in[23];
    const float* hW4   = (const float*)d_in[24];
    const float* hb4   = (const float*)d_in[25];
    float* out = (float*)d_out;

    const long long LCCn = (long long)LL * CC * CC;
    const long long LCFn = (long long)LL * CC * FF;
    const long long MCn  = (long long)MTOK * CC;
    const long long MFn  = (long long)MTOK * FF;
    const long long SCn  = (long long)BSZ * HH * TT * TT;

    char* w = (char*)d_ws;
    size_t off = 0;
    auto carve = [&](size_t bytes) -> void* {
        void* p = w + off;
        off += (bytes + 255) & ~(size_t)255;
        return p;
    };

    // weights stored TRANSPOSED: [out_features][in_features] f16
    half_t* wq16 = (half_t*)carve(LCCn * 2);
    half_t* wk16 = (half_t*)carve(LCCn * 2);
    half_t* wv16 = (half_t*)carve(LCCn * 2);
    half_t* wo16 = (half_t*)carve(LCCn * 2);
    half_t* w116 = (half_t*)carve(LCFn * 2);   // [F][C] per layer
    half_t* w216 = (half_t*)carve(LCFn * 2);   // [C][F] per layer
    float*  x32  = (float*) carve(MCn * 4);
    half_t* x16  = (half_t*)carve(MCn * 2);
    half_t* q16  = (half_t*)carve(MCn * 2);
    half_t* k16  = (half_t*)carve(MCn * 2);
    half_t* v16  = (half_t*)carve(MCn * 2);
    half_t* o16  = (half_t*)carve(MCn * 2);
    float*  tmp32 = (float*)carve(MCn * 4);
    half_t* ff16 = (half_t*)carve(MFn * 2);
    float*  sc32 = (float*) carve(SCn * 4);
    half_t* p16  = (half_t*)carve(SCn * 2);

    // ---- transpose+convert weights to f16 [N][K] ----
    cvt_tr_f16<<<dim3(CC / 32, CC / 32, LL), 256, 0, stream>>>(Wq, wq16, CC, CC);
    cvt_tr_f16<<<dim3(CC / 32, CC / 32, LL), 256, 0, stream>>>(Wk, wk16, CC, CC);
    cvt_tr_f16<<<dim3(CC / 32, CC / 32, LL), 256, 0, stream>>>(Wv, wv16, CC, CC);
    cvt_tr_f16<<<dim3(CC / 32, CC / 32, LL), 256, 0, stream>>>(Wo, wo16, CC, CC);
    cvt_tr_f16<<<dim3(FF / 32, CC / 32, LL), 256, 0, stream>>>(W1, w116, CC, FF);
    cvt_tr_f16<<<dim3(CC / 32, FF / 32, LL), 256, 0, stream>>>(W2, w216, FF, CC);

    // ---- embeddings ----
    embed_kernel<<<MTOK, 256, 0, stream>>>(idx, tok, pos, x32, x16);

    auto gemm = [&](const half_t* A, const half_t* B, float* Cf, half_t* Ch,
                    const float* bias, int M, int N, int K,
                    int lda, int ldb, int ldc,
                    long long aB, long long aH, long long bB, long long bH,
                    long long cB, long long cH, int hb,
                    int bTrans, int relu, float alpha, int batches) {
        dim3 g((N + 127) / 128, (M + 127) / 128, batches);
        gemm_wmma_f16<<<g, 256, 0, stream>>>(A, B, Cf, Ch, bias, M, N, K,
                                             lda, ldb, ldc, aB, aH, bB, bH,
                                             cB, cH, hb, bTrans, relu, alpha);
    };

    const long long TC = (long long)TT * CC;
    const long long TTn = (long long)TT * TT;

    for (int l = 0; l < LL; ++l) {
        const half_t* wq_l = wq16 + (long long)l * CC * CC;
        const half_t* wk_l = wk16 + (long long)l * CC * CC;
        const half_t* wv_l = wv16 + (long long)l * CC * CC;
        const half_t* wo_l = wo16 + (long long)l * CC * CC;
        const half_t* w1_l = w116 + (long long)l * CC * FF;
        const half_t* w2_l = w216 + (long long)l * FF * CC;

        // QKV projections: [16384,768] x [768,768]^T-stored -> f16
        gemm(x16, wq_l, nullptr, q16, nullptr, MTOK, CC, CC, CC, CC, CC,
             0, 0, 0, 0, 0, 0, 1, 1, 0, 1.0f, 1);
        gemm(x16, wk_l, nullptr, k16, nullptr, MTOK, CC, CC, CC, CC, CC,
             0, 0, 0, 0, 0, 0, 1, 1, 0, 1.0f, 1);
        gemm(x16, wv_l, nullptr, v16, nullptr, MTOK, CC, CC, CC, CC, CC,
             0, 0, 0, 0, 0, 0, 1, 1, 0, 1.0f, 1);

        // scores: per (b,h): S = Q[512x64] * K^T, alpha = 1/8
        gemm(q16, k16, sc32, nullptr, nullptr, TT, TT, HSZ, CC, CC, TT,
             TC, HSZ, TC, HSZ, (long long)HH * TTn, TTn, HH,
             1 /*K stored [s][d] == [N][K]*/, 0, 0.125f, BSZ * HH);

        // softmax rows -> f16 probabilities
        softmax_kernel<<<BSZ * HH * TT, 256, 0, stream>>>(sc32, p16);

        // O = P[512x512] * V[512x64] -> o16 in [b,t,h*64+d] layout
        gemm(p16, v16, nullptr, o16, nullptr, TT, HSZ, TT, TT, CC, CC,
             (long long)HH * TTn, TTn, TC, HSZ, TC, HSZ, HH,
             0 /*scalar staged, N=64*/, 0, 1.0f, BSZ * HH);

        // sa = O @ Wo + bo  (f32)
        gemm(o16, wo_l, tmp32, nullptr, bo + (long long)l * CC,
             MTOK, CC, CC, CC, CC, CC, 0, 0, 0, 0, 0, 0, 1, 1, 0, 1.0f, 1);

        // x = x + LN(sa)
        ln_kernel<<<MTOK, 256, 0, stream>>>(x32, x16, tmp32,
                                            ln1g + (long long)l * CC,
                                            ln1b + (long long)l * CC, 1);

        // ff1 = relu(x @ W1 + b1)  -> f16 [16384,3072]; W1T is [F][C]
        gemm(x16, w1_l, nullptr, ff16, b1 + (long long)l * FF,
             MTOK, FF, CC, CC, CC, FF, 0, 0, 0, 0, 0, 0, 1, 1, 1, 1.0f, 1);

        // ff2 = ff1 @ W2 + b2 (f32); W2T is [C][F]
        gemm(ff16, w2_l, tmp32, nullptr, b2 + (long long)l * CC,
             MTOK, CC, FF, FF, FF, CC, 0, 0, 0, 0, 0, 0, 1, 1, 0, 1.0f, 1);

        // x = x + LN(ff2)
        ln_kernel<<<MTOK, 256, 0, stream>>>(x32, x16, tmp32,
                                            ln2g + (long long)l * CC,
                                            ln2b + (long long)l * CC, 1);
    }

    // final LN (in place)
    ln_kernel<<<MTOK, 256, 0, stream>>>(x32, x16, x32, lnfg, lnfb, 0);

    // mean pool + head MLP + sigmoid
    pool_head_kernel<<<BSZ, 256, 0, stream>>>(x32, hW1, hb1, hW2, hb2,
                                              hW3, hb3, hW4, hb4, out);
}
